// MultiHeadSelfAttention_74302934221362
// MI455X (gfx1250) — matrix operand order repaired and checked by reading, not compiled, not run on hardware
//
#include <hip/hip_runtime.h>

// Problem constants (match reference)
#define BB 2
#define SS 2048
#define DD 1024
#define HH 16
#define HDIM 64
#define NTOK (BB * SS) // 4096

typedef __attribute__((ext_vector_type(16))) __bf16 v16bf;
typedef __attribute__((ext_vector_type(8)))  __bf16 v8bf;
typedef __attribute__((ext_vector_type(8)))  float  v8f;

__device__ __forceinline__ __bf16 f2bf(float f) {
  unsigned u = __builtin_bit_cast(unsigned, f);
  u += 0x7fffu + ((u >> 16) & 1u);          // round-to-nearest-even
  unsigned short h = (unsigned short)(u >> 16);
  return __builtin_bit_cast(__bf16, h);
}

__device__ __forceinline__ v8f wmma_bf16(v16bf a, v16bf b, v8f c) {
  // D = A(16x32 bf16) * B(32x16 bf16) + C(16x16 f32)
  return __builtin_amdgcn_wmma_f32_16x16x32_bf16(
      /*neg_a=*/false, a, /*neg_b=*/false, b,
      /*c_mod=*/(short)0, c, /*reuse_a=*/false, /*reuse_b=*/false);
}

__device__ __forceinline__ v16bf pack16(v8bf lo, v8bf hi) {
  v16bf r;
#pragma unroll
  for (int i = 0; i < 8; ++i) { r[i] = lo[i]; r[8 + i] = hi[i]; }
  return r;
}

// CDNA5 async global -> LDS copy, 16 bytes per lane (ASYNCcnt tracked).
// VDST = 32-bit LDS byte address, VADDR = 64-bit global address.
__device__ __forceinline__ void async_ld16(void* lds_ptr, const void* gptr) {
  unsigned l = (unsigned)(unsigned long long)lds_ptr; // low 32b = LDS offset
  asm volatile("global_load_async_to_lds_b128 %0, %1, off"
               :: "v"(l), "v"(gptr) : "memory");
}
__device__ __forceinline__ void wait_async0() {
  asm volatile("s_wait_asynccnt 0x0" ::: "memory");
}

// ---------------------------------------------------------------------------
// Kernel 0a: fp32 -> bf16 conversion (streaming, for X)
// ---------------------------------------------------------------------------
__global__ void cvt_f32_to_bf16_kernel(const float* __restrict__ in,
                                       __bf16* __restrict__ out, int n) {
  int i = blockIdx.x * blockDim.x + threadIdx.x;
  int stride = gridDim.x * blockDim.x;
  for (; i < n; i += stride) out[i] = f2bf(in[i]);
}

// ---------------------------------------------------------------------------
// Kernel 0b: fp32 [R][C] -> bf16 transposed [C][R] (for weights).
// 32x32 tiles through LDS; coalesced read and write.
// ---------------------------------------------------------------------------
__global__ __launch_bounds__(256)
void cvt_transpose_bf16_kernel(const float* __restrict__ in,
                               __bf16* __restrict__ out, int R, int C) {
  __shared__ float t[32][33];
  const int r0 = blockIdx.y * 32, c0 = blockIdx.x * 32;
  const int tid = threadIdx.x;
#pragma unroll
  for (int u = 0; u < 4; ++u) {
    int i = tid + u * 256;              // 0..1023
    int r = i >> 5, c = i & 31;
    t[r][c] = in[(size_t)(r0 + r) * C + c0 + c];
  }
  __syncthreads();
#pragma unroll
  for (int u = 0; u < 4; ++u) {
    int i = tid + u * 256;
    int c = i >> 5, r = i & 31;
    out[(size_t)(c0 + c) * R + r0 + r] = f2bf(t[r][c]);
  }
}

// ---------------------------------------------------------------------------
// Kernel 1/3: GEMM  C[MxN] = A[MxK](bf16) * B[KxN] with B given TRANSPOSED
// as Bt[N][K] (bf16).  WG = 128 threads (4 waves), WG tile 64x64,
// wave tile 32x32 (2x2 WMMA).  All tile staging via async loads to LDS.
// out_mode 0: fp32 row-major + bias      (output projection -> d_out)
// out_mode 1: bf16 [B,H,S,HDIM]          (Q/K projections)
// out_mode 2: bf16 [B,H,HDIM,S] (V^T)    (V projection)
// ---------------------------------------------------------------------------
__global__ __launch_bounds__(128)
void gemm_bf16_wmma_kernel(const __bf16* __restrict__ A,
                           const __bf16* __restrict__ Bt,
                           const float* __restrict__ bias,
                           void* __restrict__ Cout,
                           int M, int N, int K, int out_mode) {
  __shared__ __bf16 ldsA[64][32];   // [m][k] row-major
  __shared__ __bf16 ldsB[64][32];   // [n][k] (Bt is already [N][K])

  const int tid  = threadIdx.x;
  const int lane = tid & 31;
  const int wave = tid >> 5;
  const int hs   = lane >> 4;   // half-wave select
  const int ln   = lane & 15;

  const int m0 = blockIdx.y * 64;
  const int n0 = blockIdx.x * 64;
  const int wm = (wave >> 1) * 32;
  const int wn = (wave & 1) * 32;

  v8f zero = {};
  v8f acc[2][2];
  acc[0][0] = zero; acc[0][1] = zero; acc[1][0] = zero; acc[1][1] = zero;

  for (int kt = 0; kt < K; kt += 32) {
    __syncthreads();   // all waves done reading LDS from previous step
    // Stage A tile 64x32 and Bt tile 64x32: 256 16B-chunks each,
    // 2 chunks per thread per tile, no guards (tid < 128).
#pragma unroll
    for (int u = 0; u < 2; ++u) {
      int c  = tid + u * 128;         // 0..255
      int r  = c >> 2;                // row 0..63
      int cc = (c & 3) * 8;           // col 0,8,16,24
      async_ld16(&ldsA[r][cc], A  + (size_t)(m0 + r) * K + kt + cc);
      async_ld16(&ldsB[r][cc], Bt + (size_t)(n0 + r) * K + kt + cc);
    }
    if (kt + 32 < K) {
      __builtin_prefetch(A + (size_t)(m0 + (tid & 63)) * K + kt + 32, 0, 1);
    }
    wait_async0();     // this wave's async copies landed in LDS
    __syncthreads();   // everyone's copies landed

    // Fragments in contiguous k order (k = 16*hs + s), same for A and B
    v16bf afrag[2], bfrag[2];
#pragma unroll
    for (int it = 0; it < 2; ++it) {
      const __bf16* ap = &ldsA[wm + it * 16 + ln][0];
      afrag[it] = *(const v16bf*)(ap + 16 * hs);
    }
#pragma unroll
    for (int jt = 0; jt < 2; ++jt) {
      bfrag[jt] = *(const v16bf*)&ldsB[wn + jt * 16 + ln][16 * hs];
    }
#pragma unroll
    for (int it = 0; it < 2; ++it)
#pragma unroll
      for (int jt = 0; jt < 2; ++jt)
        acc[it][jt] = wmma_bf16(afrag[it], bfrag[jt], acc[it][jt]);
  }

  // --- epilogue ---
#pragma unroll
  for (int it = 0; it < 2; ++it) {
#pragma unroll
    for (int jt = 0; jt < 2; ++jt) {
      int col = n0 + wn + jt * 16 + ln;
#pragma unroll
      for (int v = 0; v < 8; ++v) {
        int row = m0 + wm + it * 16 + v + 8 * hs;   // token index
        float val = acc[it][jt][v];
        if (out_mode == 0) {
          ((float*)Cout)[(size_t)row * N + col] = val + bias[col];
        } else {
          int b = row >> 11, s = row & (SS - 1);    // SS = 2048
          int h = col >> 6,  hd = col & (HDIM - 1);
          size_t off;
          if (out_mode == 1)   // [B,H,S,HDIM]
            off = (((size_t)b * HH + h) * SS + s) * HDIM + hd;
          else                 // [B,H,HDIM,S]  (V transposed per head)
            off = (((size_t)b * HH + h) * HDIM + hd) * SS + s;
          ((__bf16*)Cout)[off] = f2bf(val);
        }
      }
    }
  }
}

// ---------------------------------------------------------------------------
// Kernel 2: flash attention.  One WG per (b*H+h, 64-query tile).
// 4 waves; each wave owns 16 query rows with online softmax.
// K tile and (already transposed) V tile staged with async loads.
// ---------------------------------------------------------------------------
__global__ __launch_bounds__(128)
void flash_attn_bf16_kernel(const __bf16* __restrict__ Qb,
                            const __bf16* __restrict__ Kb,
                            const __bf16* __restrict__ Vt,
                            __bf16* __restrict__ Hcat) {
  __shared__ __bf16 ldsK[32][64];      // [tok][hd]
  __shared__ __bf16 ldsVt[64][32];     // [hd][tok]
  __shared__ __bf16 ldsP[4][16][32];   // per-wave probs C-layout -> A-layout

  const int tid  = threadIdx.x;
  const int lane = tid & 31;
  const int wave = tid >> 5;
  const int hs   = lane >> 4;
  const int ln   = lane & 15;

  const int bh = blockIdx.y;                       // b*H + h
  const size_t base  = (size_t)bh * SS * HDIM;     // K base  [B,H,S,HDIM]
  const size_t baseT = (size_t)bh * HDIM * SS;     // Vt base [B,H,HDIM,S]
  const int q0 = blockIdx.x * 64 + wave * 16;

  // Q fragments (16 rows x 64 hd): one A-frag per 32-wide hd half.
  v16bf aQ[2];
#pragma unroll
  for (int kk = 0; kk < 2; ++kk) {
    const __bf16* qp = Qb + base + (size_t)(q0 + ln) * HDIM + kk * 32;
    aQ[kk] = *(const v16bf*)(qp + 16 * hs);
  }

  v8f zero = {};
  v8f accO[4] = {zero, zero, zero, zero};          // 16x64 output accumulator
  float mrun[8], lrun[8];
#pragma unroll
  for (int v = 0; v < 8; ++v) { mrun[v] = -1e30f; lrun[v] = 0.0f; }

  const float scale = 0.125f;                      // 1/sqrt(64)

  for (int kt = 0; kt < SS; kt += 32) {
    __syncthreads();
    // Stage K tile (32x64) and Vt tile (64x32): 256 chunks each, 2/thread.
#pragma unroll
    for (int u = 0; u < 2; ++u) {
      int c = tid + u * 128;                       // 0..255
      int t = c >> 3, d  = (c & 7) * 8;            // K:  [t][d]
      async_ld16(&ldsK[t][d], Kb + base + (size_t)(kt + t) * HDIM + d);
      int hd = c >> 2, tt = (c & 3) * 8;           // Vt: [hd][tt]
      async_ld16(&ldsVt[hd][tt], Vt + baseT + (size_t)hd * SS + kt + tt);
    }
    wait_async0();
    __syncthreads();

    // ---- scores: S = Q(16x64) @ K^T(64x32), two 16x16 tiles ----
    v8f sc[2];
#pragma unroll
    for (int jt = 0; jt < 2; ++jt) {
      v8f c = zero;
#pragma unroll
      for (int kk = 0; kk < 2; ++kk) {
        v16bf bK = *(const v16bf*)&ldsK[jt * 16 + ln][kk * 32 + 16 * hs];
        c = wmma_bf16(aQ[kk], bK, c);
      }
#pragma unroll
      for (int v = 0; v < 8; ++v) c[v] *= scale;
      sc[jt] = c;
    }

    // ---- online softmax update (row = v + 8*hs per lane) ----
#pragma unroll
    for (int v = 0; v < 8; ++v) {
      float t = fmaxf(sc[0][v], sc[1][v]);
#pragma unroll
      for (int off = 8; off > 0; off >>= 1)
        t = fmaxf(t, __shfl_xor(t, off, 32));
      float mnew = fmaxf(mrun[v], t);
      float resc = __expf(mrun[v] - mnew);
      float p0 = __expf(sc[0][v] - mnew);
      float p1 = __expf(sc[1][v] - mnew);
      sc[0][v] = p0;
      sc[1][v] = p1;
      float rs = p0 + p1;
#pragma unroll
      for (int off = 8; off > 0; off >>= 1)
        rs += __shfl_xor(rs, off, 32);
      lrun[v] = lrun[v] * resc + rs;
      mrun[v] = mnew;
#pragma unroll
      for (int jd = 0; jd < 4; ++jd) accO[jd][v] *= resc;
    }

    // ---- reshape probs C-layout -> A-layout through per-wave LDS ----
#pragma unroll
    for (int jt = 0; jt < 2; ++jt)
#pragma unroll
      for (int v = 0; v < 8; ++v)
        ldsP[wave][v + 8 * hs][jt * 16 + ln] = f2bf(sc[jt][v]);
    __syncthreads();

    v16bf aP;
    {
      const __bf16* pp = &ldsP[wave][ln][0];
      aP = *(const v16bf*)(pp + 16 * hs);
    }

    // ---- accO += P(16x32) @ V(32x64) ----
#pragma unroll
    for (int jd = 0; jd < 4; ++jd) {
      v16bf bV = *(const v16bf*)&ldsVt[jd * 16 + ln][16 * hs];
      accO[jd] = wmma_bf16(aP, bV, accO[jd]);
    }
  }

  // ---- normalize and store Hcat[B,S,D] (bf16) ----
  const int b = bh / HH, h = bh % HH;
#pragma unroll
  for (int jd = 0; jd < 4; ++jd) {
#pragma unroll
    for (int v = 0; v < 8; ++v) {
      float o = accO[jd][v] / lrun[v];
      int s = q0 + v + 8 * hs;
      size_t off = ((size_t)b * SS + s) * DD + h * HDIM + jd * 16 + ln;
      Hcat[off] = f2bf(o);
    }
  }
}

// ---------------------------------------------------------------------------
// Launch
// ---------------------------------------------------------------------------
extern "C" void kernel_launch(void* const* d_in, const int* in_sizes, int n_in,
                              void* d_out, int out_size, void* d_ws,
                              size_t ws_size, hipStream_t stream) {
  (void)in_sizes; (void)n_in; (void)out_size; (void)ws_size;

  const float* X   = (const float*)d_in[0];
  const float* W_q = (const float*)d_in[1];
  const float* W_k = (const float*)d_in[2];
  const float* W_v = (const float*)d_in[3];
  const float* W_h = (const float*)d_in[4];
  const float* b_h = (const float*)d_in[5];

  // Workspace layout (bf16 elements)
  __bf16* ws = (__bf16*)d_ws;
  const size_t N_X = (size_t)NTOK * DD;    // 4,194,304
  const size_t N_W = (size_t)DD * DD;      // 1,048,576
  __bf16* Xb   = ws;                 // [NTOK, D]
  __bf16* WqbT = Xb + N_X;           // [D, D] transposed (=[N][K])
  __bf16* WkbT = WqbT + N_W;
  __bf16* WvbT = WkbT + N_W;
  __bf16* WhbT = WvbT + N_W;
  __bf16* Qb   = WhbT + N_W;         // [B,H,S,HDIM]
  __bf16* Kb   = Qb + N_X;           // [B,H,S,HDIM]
  __bf16* Vtb  = Kb + N_X;           // [B,H,HDIM,S]
  __bf16* Hb   = Vtb + N_X;          // [B,S,D]
  // total 25,165,824 bf16 = 48 MB

  // 0) fp32 -> bf16 (X streaming; weights transposed through LDS tiles)
  cvt_f32_to_bf16_kernel<<<2048, 256, 0, stream>>>(X, Xb, (int)N_X);
  dim3 tgrid(DD / 32, DD / 32);      // (32, 32)
  cvt_transpose_bf16_kernel<<<tgrid, 256, 0, stream>>>(W_q, WqbT, DD, DD);
  cvt_transpose_bf16_kernel<<<tgrid, 256, 0, stream>>>(W_k, WkbT, DD, DD);
  cvt_transpose_bf16_kernel<<<tgrid, 256, 0, stream>>>(W_v, WvbT, DD, DD);
  cvt_transpose_bf16_kernel<<<tgrid, 256, 0, stream>>>(W_h, WhbT, DD, DD);

  // 1) Q/K/V projections: [4096,1024] x [1024,1024]
  dim3 ggrid(DD / 64, NTOK / 64);    // (16, 64)
  gemm_bf16_wmma_kernel<<<ggrid, 128, 0, stream>>>(
      Xb, WqbT, nullptr, (void*)Qb, NTOK, DD, DD, /*out_mode=*/1);
  gemm_bf16_wmma_kernel<<<ggrid, 128, 0, stream>>>(
      Xb, WkbT, nullptr, (void*)Kb, NTOK, DD, DD, 1);
  gemm_bf16_wmma_kernel<<<ggrid, 128, 0, stream>>>(
      Xb, WvbT, nullptr, (void*)Vtb, NTOK, DD, DD, /*out_mode=*/2);

  // 2) flash attention -> Hcat bf16 [B,S,D]
  dim3 agrid(SS / 64, BB * HH);      // (32, 32)
  flash_attn_bf16_kernel<<<agrid, 128, 0, stream>>>(Qb, Kb, Vtb, Hb);

  // 3) output projection + bias -> fp32 d_out [B,S,D]
  gemm_bf16_wmma_kernel<<<ggrid, 128, 0, stream>>>(
      Hb, WhbT, b_h, d_out, NTOK, DD, DD, /*out_mode=*/0);
}
